// ResidualGNN_86328842649641
// MI455X (gfx1250) — matrix operand order repaired and checked by reading, not compiled, not run on hardware
//
#include <hip/hip_runtime.h>
#include <hip/hip_bf16.h>

// ---------------------------------------------------------------------------
// ResidualGNN (TAGConv K=2, depth 4) for MI455X / gfx1250.
// GEMMs: bf16 WMMA (v_wmma_f32_16x16x32_bf16, f32 accumulate), 128x128 block
// tile, double-buffered LDS staging with fp32->bf16 conversion (packed
// v_cvt_pk_bf16_f32 when available), fragments loaded as ds_load_b128 per
// the CDNA5 16-bit A/B VGPR layouts.
// Edge propagation: one wave32 per edge, float4 gathers + f32 global atomics.
// BatchNorm: per-channel atomic reduction + fused normalize/LeakyReLU.
// ---------------------------------------------------------------------------

typedef __attribute__((ext_vector_type(16))) __bf16 v16bf;
typedef __attribute__((ext_vector_type(2)))  __bf16 v2bf;
typedef __attribute__((ext_vector_type(8)))  float  v8f;

#define HID    256
#define CIN    128
#define COUT   128
#define EPSBN  1e-5f
#define SLOPE  0.01f

// ---- helpers --------------------------------------------------------------

__device__ __forceinline__ unsigned int pack2_bf16(float lo, float hi) {
#if __has_builtin(__builtin_amdgcn_cvt_pk_bf16_f32)
  v2bf r = __builtin_amdgcn_cvt_pk_bf16_f32(lo, hi);   // single VALU op
  return __builtin_bit_cast(unsigned int, r);
#else
  unsigned int ulo = __builtin_bit_cast(unsigned int, lo);
  unsigned int uhi = __builtin_bit_cast(unsigned int, hi);
  ulo = (ulo + 0x7FFFu + ((ulo >> 16) & 1u)) >> 16;    // round-to-nearest-even
  uhi = (uhi + 0x7FFFu + ((uhi >> 16) & 1u)) >> 16;
  return (uhi << 16) | ulo;
#endif
}

__device__ __forceinline__ void atomAddF(float* p, float v) {
  unsafeAtomicAdd(p, v);   // global_atomic_add_f32 on gfx1250
}

// ---- WMMA GEMM:  C[M,Nc] (+)= A[M,K] @ W[K,Nc] (+ bias) -------------------
// flags bit0: add bias     flags bit1: accumulate into existing C
// Block tile 128x128, 8 waves (4x2), wave tile 32x64 = 2x4 WMMA tiles.
// K is processed in 32-deep steps with double-buffered LDS.

#define TM 128
#define TN 128
#define LDSPITCH 40   // bf16 elements per tile row (32 + 8 pad), 80 B

__global__ __launch_bounds__(256)
void gemm_wmma_bf16(const float* __restrict__ A, const float* __restrict__ W,
                    const float* __restrict__ bias, float* __restrict__ C,
                    int M, int K, int Nc, int flags)
{
  // A tile: row-major bf16 [row][k]   (rows of 32 valid + pad)
  // B tile: column-major bf16 [col][k] so B fragments are K-contiguous
  __shared__ __align__(16) unsigned short As[2][TM][LDSPITCH];
  __shared__ __align__(16) unsigned short Bs[2][TN][LDSPITCH];

  const int tid    = threadIdx.x;
  const int lane   = tid & 31;
  const int wave   = tid >> 5;
  const int half16 = (lane >> 4) & 1;   // lanes 0-15 vs 16-31
  const int l16    = lane & 15;

  const int mBlock = blockIdx.x * TM;
  const int nBlock = blockIdx.y * TN;
  const int wm = (wave >> 1) * 32;      // wave row offset: 0,32,64,96
  const int wn = (wave & 1) * 64;       // wave col offset: 0,64

  // staging thread mapping
  const int arow0 = tid >> 4;           // A: row = arow0 + 16*it
  const int apair = tid & 15;           // A: K-pair index (u32)
  const int bn    = tid & 127;          // B: column
  const int bkp0  = tid >> 7;           // B: K-pair = bkp0 + 2*it

  v8f    acc[2][4] = {};
  float2 aR[8];
  float  bR0[8], bR1[8];

  const int KT = K >> 5;                // number of 32-deep K steps

  auto load_regs = [&](int kt) {
    const int k0 = kt << 5;
#pragma unroll
    for (int it = 0; it < 8; ++it) {
      int r  = arow0 + it * 16;
      int gm = mBlock + r;
      if (gm >= M) gm = M - 1;          // clamp: branch-free OOB handling
      const float* ap = A + (size_t)gm * K + k0 + 2 * apair;
      aR[it] = *(const float2*)ap;
      if (it == 0 && k0 + 64 < K)       // pull A stream 2 K-tiles ahead
        __builtin_prefetch(ap + 64, 0, 0);
    }
#pragma unroll
    for (int it = 0; it < 8; ++it) {
      int kp = bkp0 + it * 2;
      const float* wp = W + (size_t)(k0 + 2 * kp) * Nc + nBlock + bn;
      bR0[it] = wp[0];
      bR1[it] = wp[Nc];
    }
  };

  auto store_lds = [&](int buf) {
#pragma unroll
    for (int it = 0; it < 8; ++it) {
      int r = arow0 + it * 16;
      ((unsigned int*)As[buf][r])[apair] = pack2_bf16(aR[it].x, aR[it].y);
    }
#pragma unroll
    for (int it = 0; it < 8; ++it) {
      int kp = bkp0 + it * 2;
      ((unsigned int*)Bs[buf][bn])[kp] = pack2_bf16(bR0[it], bR1[it]);
    }
  };

  load_regs(0);
  store_lds(0);

  for (int kt = 0; kt < KT; ++kt) {
    __syncthreads();                    // publish LDS[buf]; retire old reads
    const int buf = kt & 1;
    if (kt + 1 < KT) load_regs(kt + 1); // prefetch next step into registers

    // fragments: two ds_load_b128 each, per CDNA5 16-bit layouts
    union Frag { v16bf v; uint4 q[2]; } afr[2], bfr[4];
#pragma unroll
    for (int ti = 0; ti < 2; ++ti) {
      // lanes 0-15: K 0-7 & 16-23 ; lanes 16-31: K 8-15 & 24-31
      const unsigned short* rp = As[buf][wm + ti * 16 + l16] + half16 * 8;
      afr[ti].q[0] = *(const uint4*)(rp);
      afr[ti].q[1] = *(const uint4*)(rp + 16);
    }
#pragma unroll
    for (int tj = 0; tj < 4; ++tj) {
      // lanes 0-15: K 0-7 & 16-23 ; lanes 16-31: K 8-15 & 24-31 (column-major tile)
      const unsigned short* cp = Bs[buf][wn + tj * 16 + l16] + half16 * 8;
      bfr[tj].q[0] = *(const uint4*)(cp);
      bfr[tj].q[1] = *(const uint4*)(cp + 16);
    }

#pragma unroll
    for (int ti = 0; ti < 2; ++ti)
#pragma unroll
      for (int tj = 0; tj < 4; ++tj)
        acc[ti][tj] = __builtin_amdgcn_wmma_f32_16x16x32_bf16(
            false, afr[ti].v, false, bfr[tj].v,
            (short)0, acc[ti][tj], false, false);

    if (kt + 1 < KT) store_lds(buf ^ 1);
  }

  // --- epilogue: bias / accumulate, f32 C layout (VGPR r -> M=r / M=8+r) ---
#pragma unroll
  for (int ti = 0; ti < 2; ++ti) {
#pragma unroll
    for (int tj = 0; tj < 4; ++tj) {
      int col  = nBlock + wn + tj * 16 + l16;
      float bv = (flags & 1) ? bias[col] : 0.f;
#pragma unroll
      for (int r = 0; r < 8; ++r) {
        int row = mBlock + wm + ti * 16 + half16 * 8 + r;
        if (row < M) {
          size_t off = (size_t)row * Nc + col;
          float v = acc[ti][tj][r] + bv;
          if (flags & 2) v += C[off];
          C[off] = v;
        }
      }
    }
  }
}

// ---- edge propagation: pout[dst] += pin[src] * w  (one wave per edge) -----

__global__ __launch_bounds__(256)
void propagate_edges(const float* __restrict__ pin, float* __restrict__ pout,
                     const int* __restrict__ src, const int* __restrict__ dst,
                     const float* __restrict__ ew, int E)
{
  int e = (int)((blockIdx.x * 256u + threadIdx.x) >> 5);
  if (e >= E) return;
  int lane = threadIdx.x & 31;
  int s = src[e];
  int d = dst[e];
  float w = ew[e];
  const float4* ps = (const float4*)(pin + (size_t)s * HID);
  float* pd = pout + (size_t)d * HID;
#pragma unroll
  for (int j = 0; j < HID / (32 * 4); ++j) {
    float4 v = ps[lane + j * 32];
    int c = (lane + j * 32) * 4;
    atomAddF(pd + c + 0, v.x * w);
    atomAddF(pd + c + 1, v.y * w);
    atomAddF(pd + c + 2, v.z * w);
    atomAddF(pd + c + 3, v.w * w);
  }
}

// ---- BatchNorm (training-mode batch stats over node dim) ------------------
// stats layout: [0,C)=sum  [C,2C)=sumsq  [2C,3C)=scale  [3C,4C)=shift

__global__ __launch_bounds__(256)
void bn_stats(const float* __restrict__ x, float* __restrict__ stats,
              int Nn, int Cc)
{
  int nch = Cc >> 8;                       // channels per thread (1 or 4)
  int rows_per = (Nn + gridDim.x - 1) / gridDim.x;
  int r0 = blockIdx.x * rows_per;
  int r1 = r0 + rows_per; if (r1 > Nn) r1 = Nn;
  float s[4]  = {0.f, 0.f, 0.f, 0.f};
  float ss[4] = {0.f, 0.f, 0.f, 0.f};
  for (int r = r0; r < r1; ++r) {
    const float* xp = x + (size_t)r * Cc + threadIdx.x;
#pragma unroll 4
    for (int i = 0; i < nch; ++i) {
      float v = xp[i << 8];
      s[i] += v; ss[i] += v * v;
    }
  }
  for (int i = 0; i < nch; ++i) {
    atomAddF(&stats[threadIdx.x + (i << 8)], s[i]);
    atomAddF(&stats[Cc + threadIdx.x + (i << 8)], ss[i]);
  }
}

__global__ void bn_finalize(float* __restrict__ stats,
                            const float* __restrict__ g,
                            const float* __restrict__ b,
                            int Cc, float invN)
{
  int c = blockIdx.x * blockDim.x + threadIdx.x;
  if (c >= Cc) return;
  float m   = stats[c] * invN;
  float var = stats[Cc + c] * invN - m * m;
  float sc  = g[c] * rsqrtf(var + EPSBN);
  stats[2 * Cc + c] = sc;
  stats[3 * Cc + c] = b[c] - m * sc;
}

__device__ __forceinline__ float lrelu(float v) {
  return v > 0.f ? v : SLOPE * v;
}

__global__ __launch_bounds__(256)
void bn_apply_lrelu(const float* __restrict__ x, float* __restrict__ y,
                    const float* __restrict__ stats, int total4, int Cc)
{
  int i = blockIdx.x * 256 + threadIdx.x;
  if (i >= total4) return;
  int cp = i % (Cc >> 2);                  // channel-quad within a row
  float4 xv = ((const float4*)x)[i];
  float4 sc = ((const float4*)(stats + 2 * Cc))[cp];
  float4 sh = ((const float4*)(stats + 3 * Cc))[cp];
  float4 r;
  r.x = lrelu(fmaf(xv.x, sc.x, sh.x));
  r.y = lrelu(fmaf(xv.y, sc.y, sh.y));
  r.z = lrelu(fmaf(xv.z, sc.z, sh.z));
  r.w = lrelu(fmaf(xv.w, sc.w, sh.w));
  ((float4*)y)[i] = r;
}

// ---------------------------------------------------------------------------

extern "C" void kernel_launch(void* const* d_in, const int* in_sizes, int n_in,
                              void* d_out, int out_size, void* d_ws, size_t ws_size,
                              hipStream_t stream)
{
  const float* x       = (const float*)d_in[0];
  const int*   ei      = (const int*)  d_in[1];   // [2,E]: src then dst
  const float* ew      = (const float*)d_in[2];
  const float* W_emb   = (const float*)d_in[3];
  const float* b_emb   = (const float*)d_in[4];
  const float* conv0_W = (const float*)d_in[5];   // [3,H,H]
  const float* conv0_b = (const float*)d_in[6];
  const float* norm_g  = (const float*)d_in[7];   // [3,H]
  const float* norm_b  = (const float*)d_in[8];
  const float* conv_W  = (const float*)d_in[9];   // [3,3,H,H]
  const float* conv_b  = (const float*)d_in[10];  // [3,H]
  const float* mlp_W1  = (const float*)d_in[11];  // [3,H,4H]
  const float* mlp_b1  = (const float*)d_in[12];  // [3,4H]
  const float* mlp_g   = (const float*)d_in[13];
  const float* mlp_bb  = (const float*)d_in[14];
  const float* mlp_W2  = (const float*)d_in[15];  // [3,4H,H]
  const float* mlp_b2  = (const float*)d_in[16];  // [3,H]
  const float* W_out   = (const float*)d_in[17];
  const float* b_out   = (const float*)d_in[18];

  const int Nn = in_sizes[0] / CIN;       // 50000
  const int E  = in_sizes[2];             // 800000
  const int H  = HID;
  const int H4 = 4 * HID;

  // workspace carve-up (fp32)
  float* ws   = (float*)d_ws;
  size_t NH   = (size_t)Nn * H;
  float* bufA = ws;                // initial h, later reused as conv scratch
  float* bufB = bufA + NH;         // z (post-BN activation)
  float* p0   = bufB + NH;         // propagate ping
  float* p1   = p0   + NH;         // propagate pong
  float* bufC = p1   + NH;         // layer-0 conv out == persistent h
  float* tbuf = bufC + NH;         // N x 4H MLP hidden
  float* stats= tbuf + (size_t)Nn * H4;   // 4 * 1024 floats

  const dim3 blk(256);
  const int  propBlocks  = (E * 32 + 255) / 256;
  const int  statsBlocks = 256;

  auto gemm = [&](const float* A, const float* W, const float* bias,
                  float* C, int M, int K, int Nc, int flags) {
    dim3 grid((M + TM - 1) / TM, Nc / TN);
    gemm_wmma_bf16<<<grid, blk, 0, stream>>>(A, W, bias, C, M, K, Nc, flags);
  };

  auto tagconv = [&](const float* in, const float* Wc, const float* bc,
                     float* out) {
    // out = in@W0 + b ; out += (A in)@W1 ; out += (A^2 in)@W2
    gemm(in, Wc, bc, out, Nn, H, H, 1);
    hipMemsetAsync(p0, 0, NH * sizeof(float), stream);
    propagate_edges<<<propBlocks, blk, 0, stream>>>(in, p0, ei, ei + E, ew, E);
    gemm(p0, Wc + (size_t)H * H, nullptr, out, Nn, H, H, 2);
    hipMemsetAsync(p1, 0, NH * sizeof(float), stream);
    propagate_edges<<<propBlocks, blk, 0, stream>>>(p0, p1, ei, ei + E, ew, E);
    gemm(p1, Wc + 2 * (size_t)H * H, nullptr, out, Nn, H, H, 2);
  };

  auto bn_lrelu = [&](const float* in, float* out, const float* g,
                      const float* b, int Cc) {
    hipMemsetAsync(stats, 0, 2 * Cc * sizeof(float), stream);
    bn_stats<<<statsBlocks, blk, 0, stream>>>(in, stats, Nn, Cc);
    bn_finalize<<<(Cc + 255) / 256, blk, 0, stream>>>(stats, g, b, Cc,
                                                      1.0f / (float)Nn);
    int total4 = (Nn * Cc) / 4;
    bn_apply_lrelu<<<(total4 + 255) / 256, blk, 0, stream>>>(in, out, stats,
                                                             total4, Cc);
  };

  // 1) h = x @ W_emb + b_emb
  gemm(x, W_emb, b_emb, bufA, Nn, CIN, H, 1);

  // 2) layer-0 TAGConv: h <- tagconv(h)   (bufA -> bufC)
  tagconv(bufA, conv0_W, conv0_b, bufC);
  float* h       = bufC;
  float* scratch = bufA;   // bufA is now free; reuse as per-layer conv out

  // 3) residual blocks
  for (int l = 0; l < 3; ++l) {
    bn_lrelu(h, bufB, norm_g + (size_t)l * H, norm_b + (size_t)l * H, H);
    tagconv(bufB, conv_W + (size_t)l * 3 * H * H, conv_b + (size_t)l * H,
            scratch);
    gemm(scratch, mlp_W1 + (size_t)l * H * H4, mlp_b1 + (size_t)l * H4,
         tbuf, Nn, H, H4, 1);
    bn_lrelu(tbuf, tbuf, mlp_g + (size_t)l * H4, mlp_bb + (size_t)l * H4, H4);
    // h += t @ W2 + b2   (accumulate + bias)
    gemm(tbuf, mlp_W2 + (size_t)l * H4 * H, mlp_b2 + (size_t)l * H,
         h, Nn, H4, H, 3);
  }

  // 4) out = h @ W_out + b_out
  gemm(h, W_out, b_out, (float*)d_out, Nn, H, COUT, 1);
}
